// InfoGraph_88613765251886
// MI455X (gfx1250) — matrix operand order repaired and checked by reading, not compiled, not run on hardware
//
#include <hip/hip_runtime.h>
#include <math.h>

// Problem constants (from reference): N=100000, G=2048, D=512
#define DD 512
#define GG 2048
#define NN 100000
#define ROWS 32          // rows of the A-matrix staged per workgroup (2 row-tiles)
#define NTHREADS 256     // 8 wave32's
#define KSTEP 32         // K depth of v_wmma_f32_16x16x32_bf16
#define LOG2_F 0.69314718055994530942f

static constexpr size_t MSZ = (size_t)DD * DD;   // elements per 512x512 weight matrix

typedef __bf16 bf16_t;
typedef __attribute__((ext_vector_type(16))) __bf16 v16bf;
typedef __attribute__((ext_vector_type(8)))  __bf16 v8bf;
typedef __attribute__((ext_vector_type(8)))  float  v8f;

union AFrag { v16bf v; v8bf h[2]; };
static_assert(sizeof(AFrag) == 32, "frag size");

// ---------------------------------------------------------------------------
// Fragment loaders.
// A (16x32 bf16, MxK): lane l -> row m=l&15, half g=l>>4.
//   VGPR0..7 hold K = 16g+{0..15}  (two contiguous 16B chunks)
// B (32x16 bf16, KxN), weights stored transposed Wt[n][k] so each lane's
//   K-run is contiguous: lane l -> col n=l&15 (+n0), half g=l>>4,
//   K = 16g+{0..15} (two contiguous 16B chunks -> global_load_b128 x2).
// ---------------------------------------------------------------------------
__device__ __forceinline__ void load_a_frag(AFrag& a, const bf16_t* sh,
                                            int r0, int k0, int lane) {
  const int m = lane & 15, g = lane >> 4;
  const bf16_t* p = sh + (r0 + m) * DD + k0 + 16 * g;
  a.h[0] = *(const v8bf*)p;
  a.h[1] = *(const v8bf*)(p + 8);
}

__device__ __forceinline__ void load_b_frag(AFrag& b, const bf16_t* Wt,
                                            int n0, int k0, int lane) {
  const int n = n0 + (lane & 15), g = lane >> 4;
  const bf16_t* p = Wt + (size_t)n * DD + k0 + 16 * g;
  b.h[0] = *(const v8bf*)p;
  b.h[1] = *(const v8bf*)(p + 8);
}

// ---------------------------------------------------------------------------
// GEMM over a 32xD A tile in LDS.  Each wave owns BOTH 16-row tiles and a
// unique 64-column slice (4 col-tiles): every B fragment fetched from
// global/L2 feeds two WMMAs and no two waves load the same B data.
// acc[r*4+t]: r in {0,1} row tile, t in {0..3} col tile.
// ---------------------------------------------------------------------------
__device__ __forceinline__ void gemm_acc(const bf16_t* shin, const bf16_t* Wt,
                                         v8f acc[8], int wave, int lane) {
  const int cbase = wave * 64;
  for (int k0 = 0; k0 < DD; k0 += KSTEP) {
    AFrag a0, a1;
    load_a_frag(a0, shin, 0,  k0, lane);
    load_a_frag(a1, shin, 16, k0, lane);
#pragma unroll
    for (int t = 0; t < 4; ++t) {
      AFrag b; load_b_frag(b, Wt, cbase + 16 * t, k0, lane);
      acc[t] = __builtin_amdgcn_wmma_f32_16x16x32_bf16(
          false, a0.v, false, b.v, (short)0, acc[t], false, false);
      acc[4 + t] = __builtin_amdgcn_wmma_f32_16x16x32_bf16(
          false, a1.v, false, b.v, (short)0, acc[4 + t], false, false);
    }
  }
}

// One Linear(D,D)+bias+ReLU layer, LDS -> LDS (bf16)
__device__ __forceinline__ void gemm_relu_layer(const bf16_t* shin, bf16_t* shout,
                                                const bf16_t* Wt, const float* bias,
                                                int wave, int lane) {
  v8f acc[8] = {};
  gemm_acc(shin, Wt, acc, wave, lane);
  const int cbase = wave * 64;
  const int nl = lane & 15, g = lane >> 4;
#pragma unroll
  for (int r = 0; r < 2; ++r) {
#pragma unroll
    for (int t = 0; t < 4; ++t) {
      const int n0 = cbase + 16 * t;
      const float bv = bias[n0 + nl];
#pragma unroll
      for (int j = 0; j < 8; ++j) {     // C layout: c[j] -> row 8g+j, col nl
        float v = acc[r * 4 + t][j] + bv;
        v = v > 0.f ? v : 0.f;
        shout[(16 * r + 8 * g + j) * DD + n0 + nl] = (bf16_t)v;
      }
    }
  }
}

// Stage a ROWSxD fp32 tile from global as bf16 into LDS (row-clamped for tail)
__device__ __forceinline__ void load_x_tile(const float* X, int row0, int maxrow,
                                            bf16_t* shx, int tid) {
  for (int i = tid; i < ROWS * DD / 4; i += NTHREADS) {
    const int r = (i * 4) / DD, c = (i * 4) % DD;
    int gr = row0 + r; if (gr >= maxrow) gr = maxrow - 1;
    const float4 f = *(const float4*)(X + (size_t)gr * DD + c);
    bf16_t* o = shx + r * DD + c;
    o[0] = (bf16_t)f.x; o[1] = (bf16_t)f.y; o[2] = (bf16_t)f.z; o[3] = (bf16_t)f.w;
  }
}

// ---------------------------------------------------------------------------
// Kernel 1: fp32 -> bf16 weight convert+transpose (Wt[n][k] = W[k][n]);
// also zeroes the accumulator doubles (fresh every launch -> deterministic).
// Workspace weight order: lW0,lW1,lW2, lWs, gW0,gW1,gW2, gWs
// ---------------------------------------------------------------------------
__global__ void ig_convert_weights(const float* __restrict__ lW, const float* __restrict__ lWs,
                                   const float* __restrict__ gW, const float* __restrict__ gWs,
                                   bf16_t* __restrict__ wsW, double* __restrict__ sums) {
  const size_t id = (size_t)blockIdx.x * NTHREADS + threadIdx.x;
  if (id < 8) sums[id] = 0.0;
  if (id >= 8 * MSZ) return;
  const int m = (int)(id / MSZ);
  const size_t r = id % MSZ;
  const int k = (int)(r / DD), n = (int)(r % DD);
  const float* src;
  if (m < 3)       src = lW + (size_t)m * MSZ;
  else if (m == 3) src = lWs;
  else if (m < 7)  src = gW + (size_t)(m - 4) * MSZ;
  else             src = gWs;
  wsW[(size_t)m * MSZ + (size_t)n * DD + k] = (bf16_t)src[(size_t)k * DD + n];
}

// ---------------------------------------------------------------------------
// Kernel 2: g_enc = FF(y), written bf16 row-major [G][D] (= B-layout for res GEMM)
// ---------------------------------------------------------------------------
__global__ void __launch_bounds__(NTHREADS)
ig_gff(const float* __restrict__ y, const float* __restrict__ gB,
       const float* __restrict__ gBs, const bf16_t* __restrict__ wsW,
       bf16_t* __restrict__ genc) {
  __shared__ alignas(16) bf16_t sh0[ROWS * DD];
  __shared__ alignas(16) bf16_t sh1[ROWS * DD];
  const int tid = threadIdx.x, lane = tid & 31, wave = tid >> 5;
  const int row0 = blockIdx.x * ROWS;
  const bf16_t* W   = wsW + 4 * MSZ;        // gW block
  const bf16_t* Wst = wsW + 7 * MSZ;        // gWs

  load_x_tile(y, row0, GG, sh0, tid);                            __syncthreads();
  gemm_relu_layer(sh0, sh1, W + 0 * MSZ, gB + 0,    wave, lane); __syncthreads();
  gemm_relu_layer(sh1, sh0, W + 1 * MSZ, gB + 512,  wave, lane); __syncthreads();
  gemm_relu_layer(sh0, sh1, W + 2 * MSZ, gB + 1024, wave, lane); __syncthreads();
  load_x_tile(y, row0, GG, sh0, tid);                            __syncthreads();

  v8f acc[8] = {};
  gemm_acc(sh0, Wst, acc, wave, lane);      // shortcut x @ Ws
  const int cbase = wave * 64;
  const int nl = lane & 15, g = lane >> 4;
#pragma unroll
  for (int r = 0; r < 2; ++r) {
#pragma unroll
    for (int t = 0; t < 4; ++t) {
#pragma unroll
      for (int j = 0; j < 8; ++j) {
        const int rr = 16 * r + 8 * g + j, cc = cbase + 16 * t + nl;
        const float v = (float)sh1[rr * DD + cc] + acc[r * 4 + t][j] + gBs[cc];
        genc[(size_t)(row0 + rr) * DD + cc] = (bf16_t)v;
      }
    }
  }
}

// ---------------------------------------------------------------------------
// Kernel 3: fused l_enc FF + (l_enc @ g_enc^T) + JSD reduction. res never hits HBM.
// ---------------------------------------------------------------------------
__global__ void __launch_bounds__(NTHREADS)
ig_main(const float* __restrict__ M, const int* __restrict__ batch,
        const float* __restrict__ lB, const float* __restrict__ lBs,
        const bf16_t* __restrict__ wsW, const bf16_t* __restrict__ genc,
        double* __restrict__ sums) {
  __shared__ alignas(16) bf16_t sh0[ROWS * DD];
  __shared__ alignas(16) bf16_t sh1[ROWS * DD];
  const int tid = threadIdx.x, lane = tid & 31, wave = tid >> 5;
  const int row0 = blockIdx.x * ROWS;

  // ---- FF chain (x lives in sh0, recycled; reloaded for the shortcut) ----
  load_x_tile(M, row0, NN, sh0, tid);                              __syncthreads();
  gemm_relu_layer(sh0, sh1, wsW + 0 * MSZ, lB + 0,    wave, lane); __syncthreads();
  gemm_relu_layer(sh1, sh0, wsW + 1 * MSZ, lB + 512,  wave, lane); __syncthreads();
  gemm_relu_layer(sh0, sh1, wsW + 2 * MSZ, lB + 1024, wave, lane); __syncthreads();
  load_x_tile(M, row0, NN, sh0, tid);                              __syncthreads();

  const int nl = lane & 15, g = lane >> 4;
  {
    v8f acc[8] = {};
    gemm_acc(sh0, wsW + 3 * MSZ, acc, wave, lane);   // shortcut x @ lWs
    const int cbase = wave * 64;
#pragma unroll
    for (int r = 0; r < 2; ++r) {
#pragma unroll
      for (int t = 0; t < 4; ++t) {
#pragma unroll
        for (int j = 0; j < 8; ++j) {   // own-region read-modify-write of h3
          const int rr = 16 * r + 8 * g + j, cc = cbase + 16 * t + nl;
          const float v = (float)sh1[rr * DD + cc] + acc[r * 4 + t][j] + lBs[cc];
          sh1[rr * DD + cc] = (bf16_t)v; // sh1 now holds l_enc (bf16)
        }
      }
    }
  }
  __syncthreads();

  // ---- fused scoring GEMM + JSD epilogue ----
  // Each wave: both 16-row tiles x a unique 256-column slice (16 col-tiles).
  const int cbase = wave * 256;

  int bi[16];
#pragma unroll
  for (int r = 0; r < 2; ++r)
#pragma unroll
    for (int j = 0; j < 8; ++j) {
      const int gi = row0 + 16 * r + 8 * g + j;
      bi[r * 8 + j] = (gi < NN) ? batch[gi] : -1;
    }

  float s1 = 0.f, s2 = 0.f, s3 = 0.f;
  for (int tt = 0; tt < 16; ++tt) {
    const int n0 = cbase + tt * 16;
    v8f c[2] = {};
#pragma unroll
    for (int kk = 0; kk < 16; ++kk) {
      AFrag b;  load_b_frag(b, genc, n0, kk * KSTEP, lane);
      AFrag a0; load_a_frag(a0, sh1, 0,  kk * KSTEP, lane);
      AFrag a1; load_a_frag(a1, sh1, 16, kk * KSTEP, lane);
      c[0] = __builtin_amdgcn_wmma_f32_16x16x32_bf16(
          false, a0.v, false, b.v, (short)0, c[0], false, false);
      c[1] = __builtin_amdgcn_wmma_f32_16x16x32_bf16(
          false, a1.v, false, b.v, (short)0, c[1], false, false);
    }
    const int gcol = n0 + nl;
#pragma unroll
    for (int r = 0; r < 2; ++r) {
#pragma unroll
      for (int j = 0; j < 8; ++j) {
        if (bi[r * 8 + j] < 0) continue;          // padded row
        const float m  = c[r][j];
        const float z  = -m;                       // softplus(-m), stable form
        const float sp = fmaxf(z, 0.f) + log1pf(__expf(-fabsf(z)));
        const float f  = sp + m - LOG2_F;          // f_neg(m)
        s1 += f;
        if (bi[r * 8 + j] == gcol) { s2 += f; s3 += (LOG2_F - sp); }
      }
    }
  }

  // ---- block reduction (sh0 is free -> double scratch) + global atomics ----
  double* red = (double*)sh0;                      // 3*256 doubles = 6 KB
  red[tid] = (double)s1; red[NTHREADS + tid] = (double)s2; red[2 * NTHREADS + tid] = (double)s3;
  __syncthreads();
  for (int off = NTHREADS / 2; off > 0; off >>= 1) {
    if (tid < off) {
      red[tid]                += red[tid + off];
      red[NTHREADS + tid]     += red[NTHREADS + tid + off];
      red[2 * NTHREADS + tid] += red[2 * NTHREADS + tid + off];
    }
    __syncthreads();
  }
  if (tid == 0) {
    atomicAdd(&sums[0], red[0]);
    atomicAdd(&sums[1], red[NTHREADS]);
    atomicAdd(&sums[2], red[2 * NTHREADS]);
  }
}

// ---------------------------------------------------------------------------
// Kernel 4: combine the three sums into the scalar result.
// ---------------------------------------------------------------------------
__global__ void ig_finalize(const double* __restrict__ sums, float* __restrict__ out) {
  const double S1 = sums[0], S2 = sums[1], S3 = sums[2];
  const double E_neg = (S1 - S2) / ((double)NN * (double)(GG - 1));
  const double E_pos = S3 / (double)NN;
  out[0] = (float)(E_neg - E_pos);
}

// ---------------------------------------------------------------------------
extern "C" void kernel_launch(void* const* d_in, const int* in_sizes, int n_in,
                              void* d_out, int out_size, void* d_ws, size_t ws_size,
                              hipStream_t stream) {
  (void)in_sizes; (void)n_in; (void)out_size; (void)ws_size;
  const float* M    = (const float*)d_in[0];
  const float* y    = (const float*)d_in[1];
  const int*   bat  = (const int*)  d_in[2];
  const float* lW   = (const float*)d_in[3];
  const float* lB   = (const float*)d_in[4];
  const float* lWs  = (const float*)d_in[5];
  const float* lBs  = (const float*)d_in[6];
  const float* gW   = (const float*)d_in[7];
  const float* gB   = (const float*)d_in[8];
  const float* gWs  = (const float*)d_in[9];
  const float* gBs  = (const float*)d_in[10];

  // workspace layout: [0,64) double sums | [256, +4MB) bf16 weights | +2MB bf16 g_enc
  char*   ws   = (char*)d_ws;
  double* sums = (double*)ws;
  bf16_t* wsW  = (bf16_t*)(ws + 256);
  bf16_t* genc = (bf16_t*)(ws + 256 + 8 * MSZ * sizeof(bf16_t));

  const int cvt_blocks = (int)((8 * MSZ + NTHREADS - 1) / NTHREADS);
  ig_convert_weights<<<cvt_blocks, NTHREADS, 0, stream>>>(lW, lWs, gW, gWs, wsW, sums);
  ig_gff<<<GG / ROWS, NTHREADS, 0, stream>>>(y, gB, gBs, wsW, genc);
  ig_main<<<(NN + ROWS - 1) / ROWS, NTHREADS, 0, stream>>>(M, bat, lB, lBs, wsW, genc, sums);
  ig_finalize<<<1, 1, 0, stream>>>(sums, (float*)d_out);
}
